// Encoder_7267084665221
// MI455X (gfx1250) — compile-verified, matrix operand order checked by reading, not run on hardware
//
#include <hip/hip_runtime.h>

#define Bsz  128
#define Tlen 2048
#define Idim 64
#define Hdim 128
#define G4   512
#define GS   516   // padded gate row stride in floats (16B-aligned rows, conflict-spread)
#define HS   136   // padded h row stride in bf16 (272B rows -> 68-dword stride, conflict-free)
#define XS   72    // padded x row stride in bf16 (144B rows -> 36-dword stride, conflict-free)

typedef __attribute__((ext_vector_type(16))) __bf16 v16bf;
typedef __attribute__((ext_vector_type(8)))  __bf16 v8bf;
typedef __attribute__((ext_vector_type(4)))  __bf16 v4bf;
typedef __attribute__((ext_vector_type(8)))  float  v8f;
typedef __attribute__((ext_vector_type(4)))  float  v4f;

__device__ __forceinline__ float fast_sig(float x) {
    return 1.0f / (1.0f + __expf(-x));
}
__device__ __forceinline__ float fast_tanh(float x) {
    return 2.0f / (1.0f + __expf(-2.0f * x)) - 1.0f;
}

// Load a 16x32 WMMA operand slice for this lane from a row-major fp32 row.
// p points at row_base + (kb*32 + laneHi*8); elements 0..7 <- K k0..k0+7,
// elements 8..15 <- K k0+16..k0+23 (CDNA5 16-bit operand lane layout).
__device__ __forceinline__ v16bf load_cvt(const float* __restrict__ p) {
    v4f a = *(const v4f*)(p);
    v4f b = *(const v4f*)(p + 4);
    v4f c = *(const v4f*)(p + 16);
    v4f d = *(const v4f*)(p + 20);
    v16bf r;
    r[0]  = (__bf16)a[0];  r[1]  = (__bf16)a[1];  r[2]  = (__bf16)a[2];  r[3]  = (__bf16)a[3];
    r[4]  = (__bf16)b[0];  r[5]  = (__bf16)b[1];  r[6]  = (__bf16)b[2];  r[7]  = (__bf16)b[3];
    r[8]  = (__bf16)c[0];  r[9]  = (__bf16)c[1];  r[10] = (__bf16)c[2];  r[11] = (__bf16)c[3];
    r[12] = (__bf16)d[0];  r[13] = (__bf16)d[1];  r[14] = (__bf16)d[2];  r[15] = (__bf16)d[3];
    return r;
}

// Gather a 16x32 bf16 A-operand tile slice for this lane from a padded LDS row.
__device__ __forceinline__ v16bf lds_a_tile(const __bf16* __restrict__ row, int k0) {
    v8bf lo = *(const v8bf*)(row + k0);
    v8bf hi = *(const v8bf*)(row + k0 + 16);
    return __builtin_shufflevector(lo, hi,
                                   0, 1, 2, 3, 4, 5, 6, 7,
                                   8, 9, 10, 11, 12, 13, 14, 15);
}

__device__ __forceinline__ v8f wmma_bf16(v16bf a, v16bf b, v8f c) {
    return __builtin_amdgcn_wmma_f32_16x16x32_bf16(false, a, false, b, (short)0, c, false, false);
}

__global__ __launch_bounds__(256, 1)
void lstm_persistent(const float* __restrict__ x,
                     const float* __restrict__ h0,
                     const float* __restrict__ c0,
                     const float* __restrict__ Wih,
                     const float* __restrict__ Whh,
                     const float* __restrict__ bih,
                     const float* __restrict__ bhh,
                     float* __restrict__ out)
{
    __shared__ __align__(16) float  s_gates[16 * GS];  // [16][512] fp32, padded
    __shared__ __align__(16) __bf16 s_h[16 * HS];      // [16][128] bf16, padded
    __shared__ __align__(16) __bf16 s_x[16 * XS];      // [16][64]  bf16, padded (next-step input)

    const int tid    = threadIdx.x;
    const int wave   = tid >> 5;
    const int lane   = tid & 31;
    const int laneLo = lane & 15;
    const int laneHi = lane >> 4;
    const int btile  = blockIdx.x;   // 0..7, 16 batch rows each

    // ---------------- one-time: weights -> registers (bf16, B-operand layout) ---------
    v16bf Bhh[4][4];   // [n-tile][k-block]  gates = h @ W_hh^T -> B[k][n] = W_hh[n][k]
    v16bf Bih[4][2];
    float bsum[4];
#pragma unroll
    for (int nt = 0; nt < 4; ++nt) {
        const int n = wave * 64 + nt * 16 + laneLo;     // gate column owned by this lane
        bsum[nt] = bih[n] + bhh[n];
        const float* wh = Whh + (size_t)n * Hdim;
#pragma unroll
        for (int kb = 0; kb < 4; ++kb)
            Bhh[nt][kb] = load_cvt(wh + kb * 32 + laneHi * 8);
        const float* wi = Wih + (size_t)n * Idim;
#pragma unroll
        for (int kb = 0; kb < 2; ++kb)
            Bih[nt][kb] = load_cvt(wi + kb * 32 + laneHi * 8);
    }

    // ---------------- one-time: state init (c in regs, h0 -> LDS bf16, x[t=0] -> LDS) --
    const int m_e = tid >> 4;          // elementwise row 0..15
    const int j0  = (tid & 15) << 3;   // elementwise col base, 8 wide
    const int b_e = btile * 16 + m_e;  // global batch row

    // cooperative x-prefetch mapping: thread covers (row r_p, cols cq..cq+3)
    const int r_p = tid >> 4;
    const int cq  = (tid & 15) << 2;
    const float* xbase = x + ((size_t)(btile * 16 + r_p) * Tlen) * Idim + cq;

    float cst[8];
    {
        const float* cp = c0 + (size_t)b_e * Hdim + j0;
        v4f clo = *(const v4f*)cp;
        v4f chi = *(const v4f*)(cp + 4);
#pragma unroll
        for (int q = 0; q < 4; ++q) { cst[q] = clo[q]; cst[q + 4] = chi[q]; }

        const float* hp = h0 + (size_t)b_e * Hdim + j0;
        v4f hlo = *(const v4f*)hp;
        v4f hhi = *(const v4f*)(hp + 4);
        v8bf hv;
#pragma unroll
        for (int q = 0; q < 4; ++q) { hv[q] = (__bf16)hlo[q]; hv[q + 4] = (__bf16)hhi[q]; }
        *(v8bf*)&s_h[m_e * HS + j0] = hv;

        // stage x[., t=0, .] into LDS as bf16
        v4f xv = *(const v4f*)xbase;
        v4bf xb;
#pragma unroll
        for (int q = 0; q < 4; ++q) xb[q] = (__bf16)xv[q];
        *(v4bf*)&s_x[r_p * XS + cq] = xb;
    }
    __syncthreads();

    const int m_a = laneLo;                  // A-operand row for this lane
    float* outHlast = out;                   // [1,B,H]
    float* outHs    = out + Bsz * Hdim;      // [B,T,H]

    float hq[8];

    // =============================== recurrent scan ===================================
    for (int t = 0; t < Tlen; ++t) {
        // ---- A operands from LDS (shared by the whole workgroup) ----
        const __bf16* hrow = &s_h[m_a * HS];
        const __bf16* xrow = &s_x[m_a * XS];
        v16bf Ax0 = lds_a_tile(xrow, 0 * 32 + laneHi * 8);
        v16bf Ax1 = lds_a_tile(xrow, 1 * 32 + laneHi * 8);
        v16bf Ah[4];
#pragma unroll
        for (int kb = 0; kb < 4; ++kb)
            Ah[kb] = lds_a_tile(hrow, kb * 32 + laneHi * 8);

        // ---- gates[16, 64] for this wave: bias + x@Wih^T + h@Whh^T ----
#pragma unroll
        for (int nt = 0; nt < 4; ++nt) {
            const float bs = bsum[nt];
            v8f acc = { bs, bs, bs, bs, bs, bs, bs, bs };
            acc = wmma_bf16(Ax0,  Bih[nt][0], acc);
            acc = wmma_bf16(Ax1,  Bih[nt][1], acc);
            acc = wmma_bf16(Ah[0], Bhh[nt][0], acc);
            acc = wmma_bf16(Ah[1], Bhh[nt][1], acc);
            acc = wmma_bf16(Ah[2], Bhh[nt][2], acc);
            acc = wmma_bf16(Ah[3], Bhh[nt][3], acc);

            const int nc = wave * 64 + nt * 16 + laneLo;
#pragma unroll
            for (int r = 0; r < 8; ++r)
                s_gates[(r + 8 * laneHi) * GS + nc] = acc[r];
        }
        __syncthreads();

        // ---- prefetch next step's input tile (latency hides under the math below) ----
        const int tn = (t + 1 < Tlen) ? (t + 1) : t;
        v4f xv = *(const v4f*)(xbase + (size_t)tn * Idim);

        // ---- elementwise LSTM cell update (each thread owns 8 fixed (m,j) cells) ----
        const float* gp = &s_gates[m_e * GS];
        float ig[8], fg[8], gg[8], og[8];
        *(v4f*)(ig)     = *(const v4f*)(gp + j0);
        *(v4f*)(ig + 4) = *(const v4f*)(gp + j0 + 4);
        *(v4f*)(fg)     = *(const v4f*)(gp + 128 + j0);
        *(v4f*)(fg + 4) = *(const v4f*)(gp + 128 + j0 + 4);
        *(v4f*)(gg)     = *(const v4f*)(gp + 256 + j0);
        *(v4f*)(gg + 4) = *(const v4f*)(gp + 256 + j0 + 4);
        *(v4f*)(og)     = *(const v4f*)(gp + 384 + j0);
        *(v4f*)(og + 4) = *(const v4f*)(gp + 384 + j0 + 4);

        v8bf hb;
#pragma unroll
        for (int q = 0; q < 8; ++q) {
            const float iv = fast_sig(ig[q]);
            const float fv = fast_sig(fg[q]);
            const float gv = fast_tanh(gg[q]);
            const float ov = fast_sig(og[q]);
            cst[q] = fv * cst[q] + iv * gv;
            hq[q]  = ov * fast_tanh(cst[q]);
            hb[q]  = (__bf16)hq[q];
        }
        *(v8bf*)&s_h[m_e * HS + j0] = hb;   // next step's A_h

        // commit prefetched x tile for step t+1 (bf16, A-layout rows)
        {
            v4bf xb;
#pragma unroll
            for (int q = 0; q < 4; ++q) xb[q] = (__bf16)xv[q];
            *(v4bf*)&s_x[r_p * XS + cq] = xb;
        }

        float* op = outHs + ((size_t)b_e * Tlen + t) * Hdim + j0;
        v4f o0 = { hq[0], hq[1], hq[2], hq[3] };
        v4f o1 = { hq[4], hq[5], hq[6], hq[7] };
        *(v4f*)op       = o0;
        *(v4f*)(op + 4) = o1;

        __syncthreads();
    }

    // ---- h_last [1,B,H] ----
    float* lp = outHlast + (size_t)b_e * Hdim + j0;
    v4f l0 = { hq[0], hq[1], hq[2], hq[3] };
    v4f l1 = { hq[4], hq[5], hq[6], hq[7] };
    *(v4f*)lp       = l0;
    *(v4f*)(lp + 4) = l1;
}

extern "C" void kernel_launch(void* const* d_in, const int* in_sizes, int n_in,
                              void* d_out, int out_size, void* d_ws, size_t ws_size,
                              hipStream_t stream) {
    const float* x   = (const float*)d_in[0];  // [B,T,I]
    const float* h0  = (const float*)d_in[1];  // [B,H]
    const float* c0  = (const float*)d_in[2];  // [B,H]
    const float* Wih = (const float*)d_in[3];  // [4H,I]
    const float* Whh = (const float*)d_in[4];  // [4H,H]
    const float* bih = (const float*)d_in[5];  // [4H]
    const float* bhh = (const float*)d_in[6];  // [4H]
    float* out = (float*)d_out;                // h_last (B*H) then hs (B*T*H)

    hipLaunchKernelGGL(lstm_persistent, dim3(Bsz / 16), dim3(256), 0, stream,
                       x, h0, c0, Wih, Whh, bih, bhh, out);
}
